// FluxDoubleStreamBlock_24627342475395
// MI455X (gfx1250) — compile-verified, matrix operand order checked by reading, not run on hardware
//
#include <hip/hip_runtime.h>

typedef __attribute__((ext_vector_type(16))) __bf16 v16bf;
typedef __attribute__((ext_vector_type(8)))  __bf16 v8bf;
typedef __attribute__((ext_vector_type(8)))  float  v8f;
typedef unsigned short u16_t;

#define HSZ   3072
#define NHEADS 24
#define HDIM  128
#define MLPD  12288
#define LIMG  1024
#define LTXT  512
#define LTOT  1536
#define ATT_SCALE 0.08838834764831845f   // 128^-0.5

// LDS row strides (elements) chosen to avoid bank conflicts while keeping
// every fragment base 16-byte aligned:
//  K strip rows: 136 * 2B = 272B = 68 dwords (not a multiple of 64 banks)
//  VsT / Ps rows: 40 * 2B =  80B = 20 dwords
#define KS_LD 136
#define VS_LD 40
#define PS_LD 40

#if __has_builtin(__builtin_amdgcn_sched_group_barrier)
#define SCHED_GROUP(mask, n) __builtin_amdgcn_sched_group_barrier((mask), (n), 0)
#else
#define SCHED_GROUP(mask, n)
#endif
// mask bits: 0x002 VALU, 0x008 MFMA/WMMA, 0x020 VMEM read, 0x100 DS read

// ---------- small helpers ----------
__device__ __forceinline__ u16_t f2bf(float f) {            // native v_cvt (RNE)
  return __builtin_bit_cast(u16_t, (__bf16)f);
}
__device__ __forceinline__ float bf2f(u16_t s) {
  return (float)__builtin_bit_cast(__bf16, s);
}

__device__ __forceinline__ v8f wmma_bf16(v16bf a, v16bf b, v8f c) {
  return __builtin_amdgcn_wmma_f32_16x16x32_bf16(false, a, false, b, (short)0, c, false, false);
}

__device__ __forceinline__ v16bf cat16(v8bf lo, v8bf hi) {
  return __builtin_shufflevector(lo, hi, 0, 1, 2, 3, 4, 5, 6, 7, 8, 9, 10, 11, 12, 13, 14, 15);
}
__device__ __forceinline__ v8bf cvt8(float4 a, float4 b) {
  v8bf o;
  o[0] = (__bf16)a.x; o[1] = (__bf16)a.y; o[2] = (__bf16)a.z; o[3] = (__bf16)a.w;
  o[4] = (__bf16)b.x; o[5] = (__bf16)b.y; o[6] = (__bf16)b.z; o[7] = (__bf16)b.w;
  return o;
}

// 16x32 A/B fragment from a row-major [rows][ld] bf16 array (global or LDS).
// lane&15 -> row; kb=(lane>>4)*8; elems 0..7 -> K kb..kb+7; 8..15 -> K kb+16..kb+23.
// All call sites keep r 16-byte aligned (ld even, k0 multiples of 8 elements).
__device__ __forceinline__ v16bf frag_bf16(const u16_t* p, int row0, int ld, int k0, int lane) {
  const u16_t* r = p + (size_t)(row0 + (lane & 15)) * (size_t)ld
                     + (size_t)(k0 + ((lane >> 4) << 3));
  v8bf lo = __builtin_bit_cast(v8bf, *(const uint4*)(r));
  v8bf hi = __builtin_bit_cast(v8bf, *(const uint4*)(r + 16));
  return cat16(lo, hi);
}
// same fragment mapping from an f32 array, converting with native bf16 cvt
__device__ __forceinline__ v16bf frag_f32(const float* p, int row0, int ld, int k0, int lane) {
  const float* r = p + (size_t)(row0 + (lane & 15)) * (size_t)ld
                     + (size_t)(k0 + ((lane >> 4) << 3));
  float4 a0 = *(const float4*)(r);
  float4 a1 = *(const float4*)(r + 4);
  float4 b0 = *(const float4*)(r + 16);
  float4 b1 = *(const float4*)(r + 20);
  return cat16(cvt8(a0, a1), cvt8(b0, b1));
}

// ---------- phase 0: silu(vec) ----------
__global__ void silu_kernel(const float* __restrict__ v, float* __restrict__ out, int n) {
  int i = blockIdx.x * blockDim.x + threadIdx.x;
  if (i < n) { float x = v[i]; out[i] = x / (1.f + __expf(-x)); }
}

// ---------- phase 1: modulation GEMV: out[j] = b[j] + silu_vec . W[j][:] ----------
__global__ __launch_bounds__(256)
void modvec_kernel(const float* __restrict__ sv, const float* __restrict__ W,
                   const float* __restrict__ b, float* __restrict__ out, int N, int K) {
  int lane = threadIdx.x & 31, wave = threadIdx.x >> 5;
  int j = blockIdx.x * 8 + wave;
  if (j >= N) return;
  const float* wr = W + (size_t)j * K;
  float s = 0.f;
  for (int k = lane; k < K; k += 32) s += sv[k] * wr[k];
#pragma unroll
  for (int m = 16; m >= 1; m >>= 1) s += __shfl_xor(s, m, 32);
  if (lane == 0) out[j] = s + b[j];
}

// ---------- phase 2/7: LayerNorm + (1+scale)*x + shift -> bf16 ----------
// rows 0..511 -> srcA (txt, mod_txt), rows 512..1535 -> srcB (img, mod_img)
__global__ __launch_bounds__(256)
void lnmod_kernel(const float* __restrict__ srcA, const float* __restrict__ srcB,
                  const float* __restrict__ modT, const float* __restrict__ modI,
                  int sh_off, int sc_off, u16_t* __restrict__ xm) {
  int rr = blockIdx.x;
  const float* src; const float* mod;
  if (rr < LTXT) { src = srcA + (size_t)rr * HSZ; mod = modT; }
  else           { src = srcB + (size_t)(rr - LTXT) * HSZ; mod = modI; }
  const float* sh = mod + sh_off; const float* sc = mod + sc_off;
  __shared__ float red[2][8];
  float sum = 0.f, sq = 0.f;
  for (int c = threadIdx.x; c < HSZ; c += 256) { float x = src[c]; sum += x; sq += x * x; }
#pragma unroll
  for (int m = 16; m >= 1; m >>= 1) { sum += __shfl_xor(sum, m, 32); sq += __shfl_xor(sq, m, 32); }
  int wave = threadIdx.x >> 5, lane = threadIdx.x & 31;
  if (lane == 0) { red[0][wave] = sum; red[1][wave] = sq; }
  __syncthreads();
  float ts = 0.f, tq = 0.f;
#pragma unroll
  for (int i = 0; i < 8; ++i) { ts += red[0][i]; tq += red[1][i]; }
  float mean = ts * (1.f / HSZ);
  float var  = tq * (1.f / HSZ) - mean * mean;
  float rs   = rsqrtf(var + 1e-6f);
  u16_t* orow = xm + (size_t)rr * HSZ;
  for (int c = threadIdx.x; c < HSZ; c += 256) {
    float x = (src[c] - mean) * rs;
    orow[c] = f2bf(x * (1.f + sc[c]) + sh[c]);
  }
}

// ---------- WMMA GEMM: out = epilogue(A[bf16,MxK] @ W^T[f32,NxK] + bias) ----------
// EPI 0: bf16 out = acc+bias                  (qkv)
// EPI 1: f32 out  = resid + gate*(acc+bias)   (proj / mlp2)
// EPI 2: bf16 out = gelu_tanh(acc+bias)       (mlp0)
template <int EPI>
__global__ __launch_bounds__(256)
void gemm_kernel(const u16_t* __restrict__ A, const float* __restrict__ W,
                 const float* __restrict__ bias, const float* __restrict__ gate,
                 const float* resid, float* outF, u16_t* __restrict__ outB,
                 int M, int N, int K, int out_ld) {
  const int lane = threadIdx.x & 31;
  const int wave = threadIdx.x >> 5;     // 8 waves
  const int wm = wave >> 1;              // 0..3
  const int wn = wave & 1;               // 0..1
  const int m0 = blockIdx.x * 128 + wm * 32;   // 32 rows / wave
  const int n0 = blockIdx.y * 128 + wn * 64;   // 64 cols / wave

  v8f acc[2][4];
#pragma unroll
  for (int i = 0; i < 2; ++i)
#pragma unroll
    for (int t = 0; t < 4; ++t)
#pragma unroll
      for (int r = 0; r < 8; ++r) acc[i][t][r] = 0.f;

  for (int kk = 0; kk < K; kk += 32) {
    if (kk + 32 < K) {   // hint next tiles into cache (global_prefetch_b8)
      __builtin_prefetch(W + (size_t)(n0 + (lane & 15)) * K + kk + 32, 0, 0);
      __builtin_prefetch(A + (size_t)(m0 + (lane & 15)) * K + kk + 32, 0, 0);
    }
    v16bf af0 = frag_bf16(A, m0,      K, kk, lane);
    v16bf af1 = frag_bf16(A, m0 + 16, K, kk, lane);
    v16bf bf[4];
#pragma unroll
    for (int t = 0; t < 4; ++t) bf[t] = frag_f32(W, n0 + t * 16, K, kk, lane);
#pragma unroll
    for (int t = 0; t < 4; ++t) {
      acc[0][t] = wmma_bf16(af0, bf[t], acc[0][t]);
      acc[1][t] = wmma_bf16(af1, bf[t], acc[1][t]);
    }
    // pipeline: batch the 20 vmem reads, then cvt VALU, then the 8-WMMA burst
    SCHED_GROUP(0x020, 20);
    SCHED_GROUP(0x002, 64);
    SCHED_GROUP(0x008, 8);
  }

#pragma unroll
  for (int i = 0; i < 2; ++i)
#pragma unroll
    for (int t = 0; t < 4; ++t) {
      int col = n0 + t * 16 + (lane & 15);
      float b = bias[col];
      float g = (EPI == 1) ? gate[col] : 0.f;
#pragma unroll
      for (int r = 0; r < 8; ++r) {
        int row = m0 + i * 16 + ((lane >> 4) << 3) + r;
        float v = acc[i][t][r] + b;
        size_t o = (size_t)row * out_ld + col;
        if (EPI == 0) {
          outB[o] = f2bf(v);
        } else if (EPI == 1) {
          outF[o] = resid[o] + g * v;
        } else {
          float u = 0.7978845608028654f * (v + 0.044715f * v * v * v);
          outB[o] = f2bf(0.5f * v * (1.f + tanhf(u)));
        }
      }
    }
}

// ---------- phase 4: in-place RMS-norm + RoPE on q,k inside qkv buffer ----------
__global__ __launch_bounds__(64)
void qkvpost_kernel(u16_t* __restrict__ qkv, const float* __restrict__ pe,
                    const float* __restrict__ img_qn, const float* __restrict__ img_kn,
                    const float* __restrict__ txt_qn, const float* __restrict__ txt_kn) {
  int l = blockIdx.x / NHEADS, h = blockIdx.x % NHEADS;
  u16_t* base = qkv + (size_t)l * (3 * HSZ) + h * HDIM;
  const float* qn = (l < LTXT) ? txt_qn : img_qn;
  const float* kn = (l < LTXT) ? txt_kn : img_kn;
  int i = threadIdx.x, d0 = 2 * i, d1 = 2 * i + 1;
  float q0 = bf2f(base[d0]),        q1 = bf2f(base[d1]);
  float k0 = bf2f(base[HSZ + d0]),  k1 = bf2f(base[HSZ + d1]);
  float qs = q0 * q0 + q1 * q1, ks = k0 * k0 + k1 * k1;
#pragma unroll
  for (int m = 16; m >= 1; m >>= 1) { qs += __shfl_xor(qs, m, 32); ks += __shfl_xor(ks, m, 32); }
  __shared__ float red[2][2];
  int wave = threadIdx.x >> 5, lane = threadIdx.x & 31;
  if (lane == 0) { red[0][wave] = qs; red[1][wave] = ks; }
  __syncthreads();
  float qr = rsqrtf((red[0][0] + red[0][1]) * (1.f / HDIM) + 1.1920929e-07f);
  float kr = rsqrtf((red[1][0] + red[1][1]) * (1.f / HDIM) + 1.1920929e-07f);
  q0 *= qr * qn[d0]; q1 *= qr * qn[d1];
  k0 *= kr * kn[d0]; k1 *= kr * kn[d1];
  const float* f = pe + ((size_t)l * 64 + i) * 4;  // [j][t]
  base[d0]       = f2bf(f[0] * q0 + f[1] * q1);
  base[d1]       = f2bf(f[2] * q0 + f[3] * q1);
  base[HSZ + d0] = f2bf(f[0] * k0 + f[1] * k1);
  base[HSZ + d1] = f2bf(f[2] * k0 + f[3] * k1);
}

// ---------- phase 5: flash attention (per head, 64 q-rows per block, 4 waves) ----------
__global__ __launch_bounds__(128)
void attn_kernel(const u16_t* __restrict__ qkv, u16_t* __restrict__ out) {
  const int h = blockIdx.y;
  const int lane = threadIdx.x & 31;
  const int wave = threadIdx.x >> 5;
  const int q0row = blockIdx.x * 64 + wave * 16;

  __shared__ unsigned Ks32[32 * (KS_LD / 2)];   // K strip: [key][d] bf16, padded rows
  __shared__ u16_t    VsT[128 * VS_LD];         // V strip transposed: [d][key], padded
  __shared__ u16_t    Ps[4][16 * PS_LD];        // per-wave P staging, padded

  // Q fragments for this wave's 16 rows (K=128 -> 4 fragments), kept in registers
  v16bf qf[4];
  {
    const u16_t* qb = qkv + (size_t)(q0row + (lane & 15)) * (3 * HSZ)
                          + h * HDIM + ((lane >> 4) << 3);
#pragma unroll
    for (int s = 0; s < 4; ++s) {
      v8bf lo = __builtin_bit_cast(v8bf, *(const uint4*)(qb + s * 32));
      v8bf hi = __builtin_bit_cast(v8bf, *(const uint4*)(qb + s * 32 + 16));
      qf[s] = cat16(lo, hi);
    }
  }

  v8f o[8];
#pragma unroll
  for (int t = 0; t < 8; ++t)
#pragma unroll
    for (int r = 0; r < 8; ++r) o[t][r] = 0.f;
  float mrun[8], lrun[8];
#pragma unroll
  for (int r = 0; r < 8; ++r) { mrun[r] = -3.0e38f; lrun[r] = 0.f; }

  const unsigned* g32 = (const unsigned*)qkv;
  for (int kt = 0; kt < LTOT; kt += 32) {
    __syncthreads();
    for (int idx = threadIdx.x; idx < 32 * 64; idx += 128) {
      int row = idx >> 6, d2 = idx & 63;  // d2 = pair of head-dims
      size_t rbase = (size_t)(kt + row) * (3 * HSZ / 2) + h * (HDIM / 2) + d2;
      Ks32[row * (KS_LD / 2) + d2] = g32[rbase + (HSZ / 2)];
      unsigned vv = g32[rbase + HSZ];
      VsT[(2 * d2) * VS_LD + row]     = (u16_t)(vv & 0xFFFFu);
      VsT[(2 * d2 + 1) * VS_LD + row] = (u16_t)(vv >> 16);
    }
    __syncthreads();

    // S = Q @ K^T : batch 16 ds_load_b128 then the 8-WMMA burst
    const u16_t* KsU = (const u16_t*)Ks32;
    v16bf kb0[4], kb1[4];
#pragma unroll
    for (int s = 0; s < 4; ++s) {
      kb0[s] = frag_bf16(KsU, 0,  KS_LD, s * 32, lane);
      kb1[s] = frag_bf16(KsU, 16, KS_LD, s * 32, lane);
    }
    v8f s0, s1;
#pragma unroll
    for (int r = 0; r < 8; ++r) { s0[r] = 0.f; s1[r] = 0.f; }
#pragma unroll
    for (int s = 0; s < 4; ++s) s0 = wmma_bf16(qf[s], kb0[s], s0);
#pragma unroll
    for (int s = 0; s < 4; ++s) s1 = wmma_bf16(qf[s], kb1[s], s1);
    SCHED_GROUP(0x100, 16);
    SCHED_GROUP(0x008, 8);

    // online softmax over this 32-key strip; stats per row within half-wave
#pragma unroll
    for (int r = 0; r < 8; ++r) {
      float a0 = s0[r] * ATT_SCALE, a1 = s1[r] * ATT_SCALE;
      float mx = fmaxf(a0, a1);
#pragma unroll
      for (int m = 8; m >= 1; m >>= 1) mx = fmaxf(mx, __shfl_xor(mx, m, 16));
      float mn = fmaxf(mrun[r], mx);
      float alpha = __expf(mrun[r] - mn);
      float p0 = __expf(a0 - mn), p1 = __expf(a1 - mn);
      float rs = p0 + p1;
#pragma unroll
      for (int m = 8; m >= 1; m >>= 1) rs += __shfl_xor(rs, m, 16);
      lrun[r] = lrun[r] * alpha + rs;
      mrun[r] = mn;
#pragma unroll
      for (int t = 0; t < 8; ++t) o[t][r] *= alpha;
      int prow = ((lane >> 4) << 3) + r;
      Ps[wave][prow * PS_LD + (lane & 15)]      = f2bf(p0);
      Ps[wave][prow * PS_LD + 16 + (lane & 15)] = f2bf(p1);
    }

    // O += P @ V  (per-wave LDS region; compiler orders via DScnt)
    v16bf pf = frag_bf16(&Ps[wave][0], 0, PS_LD, 0, lane);
    v16bf vf[8];
#pragma unroll
    for (int t = 0; t < 8; ++t) vf[t] = frag_bf16(VsT, t * 16, VS_LD, 0, lane);
#pragma unroll
    for (int t = 0; t < 8; ++t) o[t] = wmma_bf16(pf, vf[t], o[t]);
    SCHED_GROUP(0x100, 18);
    SCHED_GROUP(0x008, 8);
  }

#pragma unroll
  for (int t = 0; t < 8; ++t)
#pragma unroll
    for (int r = 0; r < 8; ++r) {
      int row = q0row + ((lane >> 4) << 3) + r;
      int col = h * HDIM + t * 16 + (lane & 15);
      out[(size_t)row * HSZ + col] = f2bf(o[t][r] / lrun[r]);
    }
}

// ---------- host side ----------
extern "C" void kernel_launch(void* const* d_in, const int* in_sizes, int n_in,
                              void* d_out, int out_size, void* d_ws, size_t ws_size,
                              hipStream_t stream) {
  const float* in_img    = (const float*)d_in[0];
  const float* in_txt    = (const float*)d_in[1];
  const float* vec       = (const float*)d_in[2];
  const float* pe        = (const float*)d_in[3];
  const float* i_mod_w   = (const float*)d_in[4];
  const float* i_mod_b   = (const float*)d_in[5];
  const float* i_qkv_w   = (const float*)d_in[6];
  const float* i_qkv_b   = (const float*)d_in[7];
  const float* i_qn      = (const float*)d_in[8];
  const float* i_kn      = (const float*)d_in[9];
  const float* i_proj_w  = (const float*)d_in[10];
  const float* i_proj_b  = (const float*)d_in[11];
  const float* i_mlp0_w  = (const float*)d_in[12];
  const float* i_mlp0_b  = (const float*)d_in[13];
  const float* i_mlp2_w  = (const float*)d_in[14];
  const float* i_mlp2_b  = (const float*)d_in[15];
  const float* t_mod_w   = (const float*)d_in[16];
  const float* t_mod_b   = (const float*)d_in[17];
  const float* t_qkv_w   = (const float*)d_in[18];
  const float* t_qkv_b   = (const float*)d_in[19];
  const float* t_qn      = (const float*)d_in[20];
  const float* t_kn      = (const float*)d_in[21];
  const float* t_proj_w  = (const float*)d_in[22];
  const float* t_proj_b  = (const float*)d_in[23];
  const float* t_mlp0_w  = (const float*)d_in[24];
  const float* t_mlp0_b  = (const float*)d_in[25];
  const float* t_mlp2_w  = (const float*)d_in[26];
  const float* t_mlp2_b  = (const float*)d_in[27];

  float* out_img = (float*)d_out;                      // [1024][3072]
  float* out_txt = out_img + (size_t)LIMG * HSZ;       // [512][3072]

  // workspace layout (~50 MB peak, phase-based reuse)
  char* ws = (char*)d_ws;
  float* silu_v  = (float*)(ws);                        // 3072 f32
  float* mod_img = (float*)(ws + (64 << 10));           // 18432 f32
  float* mod_txt = (float*)(ws + (192 << 10));          // 18432 f32
  u16_t* bufX    = (u16_t*)(ws + (1 << 20));            // 1536*3072 bf16: xm / attn_out / xm2
  u16_t* bufQH   = (u16_t*)(ws + (12u << 20));          // qkv (1536*9216) then hidden (1536*12288)

  // 0. silu(vec)
  silu_kernel<<<12, 256, 0, stream>>>(vec, silu_v, HSZ);
  // 1. modulation vectors (6*3072 each stream)
  modvec_kernel<<<2304, 256, 0, stream>>>(silu_v, i_mod_w, i_mod_b, mod_img, 6 * HSZ, HSZ);
  modvec_kernel<<<2304, 256, 0, stream>>>(silu_v, t_mod_w, t_mod_b, mod_txt, 6 * HSZ, HSZ);
  // 2. LN1 + mod (sh1 @0, sc1 @3072) -> xm (txt rows 0..511, img rows 512..1535)
  lnmod_kernel<<<LTOT, 256, 0, stream>>>(in_txt, in_img, mod_txt, mod_img, 0, HSZ, bufX);
  // 3. QKV GEMMs -> bufQH [1536][9216] bf16
  gemm_kernel<0><<<dim3(LTXT / 128, 9216 / 128), 256, 0, stream>>>(
      bufX, t_qkv_w, t_qkv_b, nullptr, nullptr, nullptr, bufQH,
      LTXT, 3 * HSZ, HSZ, 3 * HSZ);
  gemm_kernel<0><<<dim3(LIMG / 128, 9216 / 128), 256, 0, stream>>>(
      bufX + (size_t)LTXT * HSZ, i_qkv_w, i_qkv_b, nullptr, nullptr, nullptr,
      bufQH + (size_t)LTXT * 3 * HSZ, LIMG, 3 * HSZ, HSZ, 3 * HSZ);
  // 4. RMS-norm + RoPE on q,k (in place)
  qkvpost_kernel<<<LTOT * NHEADS, 64, 0, stream>>>(bufQH, pe, i_qn, i_kn, t_qn, t_kn);
  // 5. attention -> bufX [1536][3072] bf16
  attn_kernel<<<dim3(LTOT / 64, NHEADS), 128, 0, stream>>>(bufQH, bufX);
  // 6. proj + gate1 + residual -> d_out
  gemm_kernel<1><<<dim3(LTXT / 128, HSZ / 128), 256, 0, stream>>>(
      bufX, t_proj_w, t_proj_b, mod_txt + 2 * HSZ, in_txt, out_txt, nullptr,
      LTXT, HSZ, HSZ, HSZ);
  gemm_kernel<1><<<dim3(LIMG / 128, HSZ / 128), 256, 0, stream>>>(
      bufX + (size_t)LTXT * HSZ, i_proj_w, i_proj_b, mod_img + 2 * HSZ, in_img, out_img, nullptr,
      LIMG, HSZ, HSZ, HSZ);
  // 7. LN2 + mod (sh2 @9216, sc2 @12288) on d_out -> xm2 (reuses bufX)
  lnmod_kernel<<<LTOT, 256, 0, stream>>>(out_txt, out_img, mod_txt, mod_img, 3 * HSZ, 4 * HSZ, bufX);
  // 8. MLP0 + gelu -> bufQH as hidden [1536][12288] bf16
  gemm_kernel<2><<<dim3(LTXT / 128, MLPD / 128), 256, 0, stream>>>(
      bufX, t_mlp0_w, t_mlp0_b, nullptr, nullptr, nullptr, bufQH,
      LTXT, MLPD, HSZ, MLPD);
  gemm_kernel<2><<<dim3(LIMG / 128, MLPD / 128), 256, 0, stream>>>(
      bufX + (size_t)LTXT * HSZ, i_mlp0_w, i_mlp0_b, nullptr, nullptr, nullptr,
      bufQH + (size_t)LTXT * MLPD, LIMG, MLPD, HSZ, MLPD);
  // 9. MLP2 + gate2 + residual (resid aliases d_out; single RMW per element)
  gemm_kernel<1><<<dim3(LTXT / 128, HSZ / 128), 256, 0, stream>>>(
      bufQH, t_mlp2_w, t_mlp2_b, mod_txt + 5 * HSZ, out_txt, out_txt, nullptr,
      LTXT, HSZ, MLPD, HSZ);
  gemm_kernel<1><<<dim3(LIMG / 128, HSZ / 128), 256, 0, stream>>>(
      bufQH + (size_t)LTXT * MLPD, i_mlp2_w, i_mlp2_b, mod_img + 5 * HSZ, out_img, out_img, nullptr,
      LIMG, HSZ, MLPD, HSZ);
  (void)in_sizes; (void)n_in; (void)out_size; (void)ws_size;
}
